// GNN_3410204033431
// MI455X (gfx1250) — matrix operand order repaired and checked by reading, not compiled, not run on hardware
//
#include <hip/hip_runtime.h>
#include <hip/hip_bf16.h>

#define D_IN   768
#define D_HID  64
#define NCLS   7
#define GST    8     // padded stride for layer-2 feature rows
#define NCHUNK (D_IN / 128)   // 6 K-chunks of 128

typedef __attribute__((ext_vector_type(16))) __bf16 v16bf;
typedef __attribute__((ext_vector_type(8)))  float  v8f;

__device__ __forceinline__ unsigned pack_bf16(float a, float b) {
    unsigned short lo = __builtin_bit_cast(unsigned short, (__bf16)a);
    unsigned short hi = __builtin_bit_cast(unsigned short, (__bf16)b);
    return (unsigned)lo | ((unsigned)hi << 16);
}

// ---------------- utility kernels ----------------
__global__ void zero_f32(float* __restrict__ p, int n) {
    int i = blockIdx.x * blockDim.x + threadIdx.x;
    if (i < n) p[i] = 0.0f;
}

__global__ void deg_kernel(const int* __restrict__ dst, float* __restrict__ deg, int E) {
    int e = blockIdx.x * blockDim.x + threadIdx.x;
    if (e < E) unsafeAtomicAdd(&deg[dst[e]], 1.0f);
}

__global__ void dis_kernel(const float* __restrict__ deg, float* __restrict__ dis, int n) {
    int i = blockIdx.x * blockDim.x + threadIdx.x;
    if (i < n) dis[i] = rsqrtf(deg[i] + 1.0f);
}

// ---------------- layer-1 GEMM: H1 = X @ W1 (bf16 WMMA, f32 accumulate) ----------------
// Block: 256 threads = 8 waves; wave w computes rows [blk*128 + w*16, +16) x all 64 cols.
// W1 is staged per-128-K-chunk in LDS, pre-packed into B-fragment order and double-buffered:
//   dword index = ((ks*2 + half)*64 + N)*8 + r holds bf16 pair {W[K][N], W[K+1][N]},
//   K = kc + ks*32 + half*16 + 2r.   4096 dwords = 16 KB per buffer.
__global__ void __launch_bounds__(256)
gemm1_wmma(const float* __restrict__ X, const float* __restrict__ W1,
           float* __restrict__ H1, int nNodes) {
    __shared__ unsigned ldsw[2][4096];

    const int lane = threadIdx.x & 31;
    const int wave = threadIdx.x >> 5;
    const int half = lane >> 4;      // 0: lanes 0-15, 1: lanes 16-31
    const int ml   = lane & 15;
    const int row0 = blockIdx.x * 128 + wave * 16;
    const int rowM = row0 + ml;
    const int rowC = rowM < nNodes ? rowM : (nNodes - 1);
    const float* __restrict__ xrow = X + (size_t)rowC * D_IN;

    // prologue: fill buffer 0 with chunk kc = 0
    #pragma unroll
    for (int it = 0; it < 16; ++it) {
        int i  = threadIdx.x + it * 256;
        int r  = i & 7;
        int n  = (i >> 3) & 63;
        int hs = i >> 9;                               // ks*2 + half
        int K  = (hs >> 1) * 32 + (hs & 1) * 16 + 2 * r;
        ldsw[0][i] = pack_bf16(W1[K * D_HID + n], W1[(K + 1) * D_HID + n]);
    }
    __syncthreads();

    v8f acc[4] = {v8f{}, v8f{}, v8f{}, v8f{}};

    for (int kci = 0; kci < NCHUNK; ++kci) {
        const int kc  = kci * 128;
        const int buf = kci & 1;

        // stage next chunk's W1 into registers (global loads issue early,
        // their latency hides under the WMMA chain below)
        float w0[16], w1[16];
        if (kci + 1 < NCHUNK) {
            #pragma unroll
            for (int it = 0; it < 16; ++it) {
                int i  = threadIdx.x + it * 256;
                int r  = i & 7;
                int n  = (i >> 3) & 63;
                int hs = i >> 9;
                int K  = (kc + 128) + (hs >> 1) * 32 + (hs & 1) * 16 + 2 * r;
                w0[it] = W1[K * D_HID + n];
                w1[it] = W1[(K + 1) * D_HID + n];
            }
        }

        #pragma unroll
        for (int ks = 0; ks < 4; ++ks) {
            const int kb = kc + ks * 32;
            // A fragment: lane holds M=ml, K = kb + {half*16..+15}
            v16bf a;
            const int k1 = kb + half * 16;
            const int k2 = kb + half * 16 + 8;
            #pragma unroll
            for (int t = 0; t < 8; ++t) a[t]     = (__bf16)xrow[k1 + t];
            #pragma unroll
            for (int t = 0; t < 8; ++t) a[8 + t] = (__bf16)xrow[k2 + t];

            // hoist all 4 B fragments so the 8 ds_load_b128 issue as one clause
            union BU { uint4 q[2]; v16bf v; };
            BU bu[4];
            const unsigned base = (unsigned)(ks * 2 + half) * 512u + (unsigned)ml * 8u;
            #pragma unroll
            for (int t = 0; t < 4; ++t) {
                const uint4* p = (const uint4*)&ldsw[buf][base + (unsigned)t * 128u];
                bu[t].q[0] = p[0];
                bu[t].q[1] = p[1];
            }
            #pragma unroll
            for (int t = 0; t < 4; ++t)
                acc[t] = __builtin_amdgcn_wmma_f32_16x16x32_bf16(
                    false, a, false, bu[t].v, (short)0, acc[t], false, false);
        }

        // commit staged weights into the other buffer, then one barrier
        if (kci + 1 < NCHUNK) {
            #pragma unroll
            for (int it = 0; it < 16; ++it)
                ldsw[buf ^ 1][threadIdx.x + it * 256] = pack_bf16(w0[it], w1[it]);
        }
        __syncthreads();
    }

    // C layout: VGPR v -> M = v + 8*half, N = ml
    #pragma unroll
    for (int t = 0; t < 4; ++t) {
        #pragma unroll
        for (int v = 0; v < 8; ++v) {
            int row = row0 + half * 8 + v;
            if (row < nNodes)
                H1[(size_t)row * D_HID + t * 16 + ml] = acc[t][v];
        }
    }
}

// ---------------- layer-1 edge aggregation: agg1[dst] += H1[src] * norm ----------------
__global__ void agg1_kernel(const int* __restrict__ src, const int* __restrict__ dst,
                            const float* __restrict__ dis, const float* __restrict__ H1,
                            float* __restrict__ agg, int E) {
    long long t = (long long)blockIdx.x * blockDim.x + threadIdx.x;
    int e  = (int)(t >> 5);
    int j2 = (int)(t & 31);          // float2 lane within the 64-wide row
    if (e < E) {
        int s = src[e], d = dst[e];
        float norm = dis[s] * dis[d];
        float2 hv = ((const float2*)H1)[(size_t)s * 32 + j2];
        unsafeAtomicAdd(&agg[(size_t)d * D_HID + 2 * j2],     hv.x * norm);
        unsafeAtomicAdd(&agg[(size_t)d * D_HID + 2 * j2 + 1], hv.y * norm);
    }
}

// ---------------- self-loop + bias + relu (in place on agg1 -> becomes H) ----------------
__global__ void relu_self(float* __restrict__ h, const float* __restrict__ H1,
                          const float* __restrict__ dis, const float* __restrict__ b1,
                          int nElem) {
    int idx = blockIdx.x * blockDim.x + threadIdx.x;
    if (idx < nElem) {
        int i = idx >> 6, j = idx & 63;
        float dsq = dis[i] * dis[i];
        float v = h[idx] + H1[idx] * dsq + b1[j];
        h[idx] = v > 0.0f ? v : 0.0f;
    }
}

// ---------------- layer-2 GEMM: G = H @ W2 (tiny N=7, VALU is roofline-optimal) ----------
__global__ void gemm2_kernel(const float* __restrict__ H, const float* __restrict__ W2,
                             float* __restrict__ G, int nNodes) {
    __shared__ float w[D_HID * NCLS];
    for (int i = threadIdx.x; i < D_HID * NCLS; i += blockDim.x) w[i] = W2[i];
    __syncthreads();
    int i = blockIdx.x * blockDim.x + threadIdx.x;
    if (i >= nNodes) return;
    const float* hr = H + (size_t)i * D_HID;
    float acc[NCLS] = {};
    #pragma unroll 8
    for (int k = 0; k < D_HID; ++k) {
        float hv = hr[k];
        #pragma unroll
        for (int c = 0; c < NCLS; ++c) acc[c] += hv * w[k * NCLS + c];
    }
    #pragma unroll
    for (int c = 0; c < NCLS; ++c) G[(size_t)i * GST + c] = acc[c];
    G[(size_t)i * GST + NCLS] = 0.0f;
}

// ---------------- layer-2 edge aggregation ----------------
__global__ void agg2_kernel(const int* __restrict__ src, const int* __restrict__ dst,
                            const float* __restrict__ dis, const float* __restrict__ G,
                            float* __restrict__ agg, int E) {
    long long t = (long long)blockIdx.x * blockDim.x + threadIdx.x;
    int e = (int)(t >> 3);
    int c = (int)(t & 7);
    if (e < E && c < NCLS) {
        int s = src[e], d = dst[e];
        float norm = dis[s] * dis[d];
        unsafeAtomicAdd(&agg[(size_t)d * GST + c], G[(size_t)s * GST + c] * norm);
    }
}

// ---------------- self-loop + bias + log_softmax ----------------
__global__ void final_kernel(const float* __restrict__ agg2, const float* __restrict__ G,
                             const float* __restrict__ dis, const float* __restrict__ b2,
                             float* __restrict__ out, int nNodes) {
    int i = blockIdx.x * blockDim.x + threadIdx.x;
    if (i >= nNodes) return;
    float dsq = dis[i] * dis[i];
    float v[NCLS];
    float m = -1e30f;
    #pragma unroll
    for (int c = 0; c < NCLS; ++c) {
        v[c] = agg2[(size_t)i * GST + c] + G[(size_t)i * GST + c] * dsq + b2[c];
        m = v[c] > m ? v[c] : m;
    }
    float s = 0.0f;
    #pragma unroll
    for (int c = 0; c < NCLS; ++c) s += expf(v[c] - m);
    float lse = logf(s);
    #pragma unroll
    for (int c = 0; c < NCLS; ++c) out[(size_t)i * NCLS + c] = v[c] - m - lse;
}

extern "C" void kernel_launch(void* const* d_in, const int* in_sizes, int n_in,
                              void* d_out, int out_size, void* d_ws, size_t ws_size,
                              hipStream_t stream) {
    const float* X  = (const float*)d_in[0];
    const int*   ei = (const int*)d_in[1];   // harness provides integer inputs as int32
    const float* W1 = (const float*)d_in[2];
    const float* b1 = (const float*)d_in[3];
    const float* W2 = (const float*)d_in[4];
    const float* b2 = (const float*)d_in[5];
    float* out = (float*)d_out;

    const int nN = in_sizes[0] / D_IN;
    const int E  = in_sizes[1] / 2;
    const int* src = ei;
    const int* dst = ei + E;

    // workspace layout (all float, contiguous): nN*(1+1+64+64+8+8) floats ≈ 58 MB
    float* deg  = (float*)d_ws;
    float* dis  = deg  + (size_t)nN;
    float* H1   = dis  + (size_t)nN;
    float* agg1 = H1   + (size_t)nN * D_HID;   // becomes H after relu_self
    float* G    = agg1 + (size_t)nN * D_HID;
    float* agg2 = G    + (size_t)nN * GST;

    const int B = 256;
    zero_f32  <<<(nN + B - 1) / B, B, 0, stream>>>(deg, nN);
    deg_kernel<<<(E + B - 1) / B, B, 0, stream>>>(dst, deg, E);
    dis_kernel<<<(nN + B - 1) / B, B, 0, stream>>>(deg, dis, nN);

    gemm1_wmma<<<(nN + 127) / 128, 256, 0, stream>>>(X, W1, H1, nN);

    zero_f32  <<<(nN * D_HID + B - 1) / B, B, 0, stream>>>(agg1, nN * D_HID);
    {
        long long items = (long long)E * 32;
        agg1_kernel<<<(unsigned)((items + B - 1) / B), B, 0, stream>>>(src, dst, dis, H1, agg1, E);
    }
    relu_self<<<(nN * D_HID + B - 1) / B, B, 0, stream>>>(agg1, H1, dis, b1, nN * D_HID);

    gemm2_kernel<<<(nN + B - 1) / B, B, 0, stream>>>(agg1, W2, G, nN);

    zero_f32<<<(nN * GST + B - 1) / B, B, 0, stream>>>(agg2, nN * GST);
    {
        long long items = (long long)E * GST;
        agg2_kernel<<<(unsigned)((items + B - 1) / B), B, 0, stream>>>(src, dst, dis, G, agg2, E);
    }
    final_kernel<<<(nN + B - 1) / B, B, 0, stream>>>(agg2, G, dis, b2, out, nN);
}
